// sub_Attention_47132971107011
// MI455X (gfx1250) — matrix operand order repaired and checked by reading, not compiled, not run on hardware
//
#include <hip/hip_runtime.h>
#include <hip/hip_bf16.h>
#include <math.h>

#define DIM 128
#define HEAD_DIM 32
#define NSP 4096
#define IMG 512
#define NPIX (IMG*IMG)
#define NEGV (-9000000000000000.0f)
#define TEMPSCALE 0.08838834764831845f  // 128^-0.5

typedef __attribute__((ext_vector_type(16))) __bf16          bf16x16;
typedef __attribute__((ext_vector_type(8)))  float           f32x8;
typedef __attribute__((ext_vector_type(16))) unsigned short  u16x16;
typedef __attribute__((ext_vector_type(8)))  unsigned short  u16x8;

__device__ inline unsigned short f32_to_bf16(float f) {
  unsigned u = __float_as_uint(f);
  u += 0x7FFFu + ((u >> 16) & 1u);          // round-to-nearest-even
  return (unsigned short)(u >> 16);
}
__device__ inline bf16x16 as_bf16x16(u16x16 u) { return __builtin_bit_cast(bf16x16, u); }

// ---------------------------------------------------------------- adjacency
__global__ __launch_bounds__(256) void zero_u32(unsigned* __restrict__ p, int n) {
  int t = blockIdx.x * 256 + threadIdx.x;
  if (t < n) p[t] = 0u;
}

__global__ __launch_bounds__(256) void build_adj(const int* __restrict__ seg,
                                                 unsigned* __restrict__ adj) {
  int t = blockIdx.x * 256 + threadIdx.x;
  if (t >= 510 * 510) return;
  int i = t / 510, j = t % 510;
  int a = seg[i * IMG + j],       b = seg[i * IMG + j + 1];
  int c = seg[(i + 1) * IMG + j], d = seg[(i + 1) * IMG + j + 1];
  int mx = max(max(a, b), max(c, d));
  int mn = min(min(a, b), min(c, d));
  if (mx != mn) {
    atomicOr(&adj[(size_t)mx * 128 + (mn >> 5)], 1u << (mn & 31));
    atomicOr(&adj[(size_t)mn * 128 + (mx >> 5)], 1u << (mx & 31));
  }
}

// ---------------------------------------------------------------- qkv projection
// q0raw/k0raw: f32 [4096][32] (head 0 only).  vT: bf16 [128][4096] (transposed for
// contiguous WMMA B-fragment loads).
__global__ __launch_bounds__(256) void qkv_kernel(const float* __restrict__ spf,
                                                  const float* __restrict__ w_qkv,
                                                  float* __restrict__ q0raw,
                                                  float* __restrict__ k0raw,
                                                  unsigned short* __restrict__ vT) {
  int t = blockIdx.x * 256 + threadIdx.x;
  if (t >= NSP * 192) return;
  int n = t & (NSP - 1);
  int j = t >> 12;
  int c;
  if (j < 32)      c = j;             // q head 0
  else if (j < 64) c = 128 + (j - 32);// k head 0
  else             c = 256 + (j - 64);// v (all heads, concatenated)
  float acc = 0.f;
#pragma unroll 4
  for (int d = 0; d < DIM; ++d)
    acc += spf[(size_t)d * NSP + n] * w_qkv[d * 384 + c];
  if (j < 32)      q0raw[n * 32 + j] = acc;
  else if (j < 64) k0raw[n * 32 + (j - 32)] = acc;
  else             vT[(size_t)(j - 64) * NSP + n] = f32_to_bf16(acc);
}

// ---------------------------------------------------------------- L2 normalize q0/k0 -> bf16
__global__ __launch_bounds__(256) void norm_kernel(const float* __restrict__ q0raw,
                                                   const float* __restrict__ k0raw,
                                                   unsigned short* __restrict__ q0n,
                                                   unsigned short* __restrict__ k0n) {
  int wid  = (blockIdx.x * 256 + threadIdx.x) >> 5;  // one wave32 per 32-dim row
  int lane = threadIdx.x & 31;
  if (wid >= 2 * NSP) return;
  int sel = wid >> 12;              // 0 = q, 1 = k
  int row = wid & (NSP - 1);
  const float* src = sel ? k0raw : q0raw;
  unsigned short* dst = sel ? k0n : q0n;
  float x = src[row * 32 + lane];
  float ss = x * x;
  ss += __shfl_xor(ss, 1, 32);  ss += __shfl_xor(ss, 2, 32);
  ss += __shfl_xor(ss, 4, 32);  ss += __shfl_xor(ss, 8, 32);
  ss += __shfl_xor(ss, 16, 32);
  float denom = fmaxf(sqrtf(ss), 1e-12f);
  dst[row * 32 + lane] = f32_to_bf16(x / denom);
}

// ---------------------------------------------------------------- fused masked attention
// One wave32 per 16-row query tile.  For each 32-kv chunk:
//   2x v_wmma_f32_16x16x32_bf16 (scores) -> mask/exp -> P via LDS transpose ->
//   8x v_wmma_f32_16x16x32_bf16 (P @ V, 128 output dims).
// Row max is provably 1.0 (diagonal forced to 1.0; others <= TEMP), so p = exp(s-1).
// Adjacency bits (8 KB) are brought in with ASYNCcnt-tracked async LDS loads.
__global__ __launch_bounds__(32) void attn_kernel(const unsigned short* __restrict__ q0n,
                                                  const unsigned short* __restrict__ k0n,
                                                  const unsigned short* __restrict__ vT,
                                                  const unsigned* __restrict__ adj,
                                                  float* __restrict__ attn_out) {
  __shared__ __align__(16) unsigned       adjLDS[16 * 128];  // 8 KB: adjacency bits for 16 rows
  __shared__ __align__(32) unsigned short pLDS[16 * 32];     // 1 KB: P tile for layout transpose

  const int lane  = threadIdx.x;
  const int half  = lane >> 4;      // wave32 half: selects K/M sub-ranges per ISA layout
  const int lm    = lane & 15;
  const int qbase = blockIdx.x * 16;

  // Async fill of adjLDS: 512 x 16B, each lane issues 16 global_load_async_to_lds_b128
  // (no VGPR round-trip; tracked by ASYNCcnt, drained with s_wait_asynccnt).
  {
    const unsigned lds_base = (unsigned)(uintptr_t)(&adjLDS[0]);
    const char* gbase = (const char*)adj + (size_t)qbase * 512;
#pragma unroll 4
    for (int i = lane; i < 512; i += 32) {
      unsigned lds_addr = lds_base + (unsigned)i * 16u;
      const char* gaddr = gbase + (size_t)i * 16u;
      asm volatile("global_load_async_to_lds_b128 %0, %1, off"
                   :: "v"(lds_addr), "v"(gaddr) : "memory");
    }
    asm volatile("s_wait_asynccnt 0x0" ::: "memory");
  }

  // A-fragment of Q (16x32 bf16): lane holds K = half*16+0..15 of row lm.
  u16x8 qa0 = *(const u16x8*)(q0n + (size_t)(qbase + lm) * HEAD_DIM + half * 16);
  u16x8 qa1 = *(const u16x8*)(q0n + (size_t)(qbase + lm) * HEAD_DIM + half * 16 + 8);
  u16x16 qU;
#pragma unroll
  for (int i = 0; i < 8; i++) { qU[i] = qa0[i]; qU[8 + i] = qa1[i]; }
  const bf16x16 qfrag = as_bf16x16(qU);

  __syncthreads();

  f32x8 acc[8];
  const f32x8 czero = {0.f, 0.f, 0.f, 0.f, 0.f, 0.f, 0.f, 0.f};
#pragma unroll
  for (int i = 0; i < 8; i++) acc[i] = czero;
  float z[8];
#pragma unroll
  for (int r = 0; r < 8; r++) z[r] = 0.f;

  for (int chunk = 0; chunk < NSP / 32; ++chunk) {
    const int kvbase = chunk * 32;
    // Prefetch next chunk's K rows (32 rows x 64B == one cacheline-ish per lane).
    if (chunk + 1 < NSP / 32)
      __builtin_prefetch(k0n + (size_t)(kvbase + 32 + lane) * HEAD_DIM, 0, 1);

    // B-fragments of K^T (32x16 bf16): lane holds 16 contiguous head-dims of kv row.
    u16x16 b0 = *(const u16x16*)(k0n + (size_t)(kvbase + lm) * HEAD_DIM + half * 16);
    u16x16 b1 = *(const u16x16*)(k0n + (size_t)(kvbase + 16 + lm) * HEAD_DIM + half * 16);
    f32x8 s0 = __builtin_amdgcn_wmma_f32_16x16x32_bf16(false, qfrag, false, as_bf16x16(b0),
                                                       (short)0, czero, false, false);
    f32x8 s1 = __builtin_amdgcn_wmma_f32_16x16x32_bf16(false, qfrag, false, as_bf16x16(b1),
                                                       (short)0, czero, false, false);

#pragma unroll
    for (int r = 0; r < 8; r++) {
      const int m    = half * 8 + r;                 // C/D layout: VGPR r -> row half*8+r
      const int qrow = qbase + m;
      const unsigned w = adjLDS[m * 128 + chunk];    // one u32 == this 32-kv chunk's bits
      {
        const int col = kvbase + lm;
        float v = ((w >> lm) & 1u) ? s0[r] * TEMPSCALE : NEGV;
        if (qrow == col) v = 1.0f;
        float p = __expf(v - 1.0f);                  // row max == 1.0 exactly
        z[r] += p;
        pLDS[m * 32 + lm] = f32_to_bf16(p);
      }
      {
        const int col = kvbase + 16 + lm;
        float v = ((w >> (16 + lm)) & 1u) ? s1[r] * TEMPSCALE : NEGV;
        if (qrow == col) v = 1.0f;
        float p = __expf(v - 1.0f);
        z[r] += p;
        pLDS[m * 32 + 16 + lm] = f32_to_bf16(p);
      }
    }
    __syncthreads();
    // Re-load P in A-fragment layout (16x32 bf16).
    u16x8 p0 = *(const u16x8*)(pLDS + lm * 32 + half * 16);
    u16x8 p1 = *(const u16x8*)(pLDS + lm * 32 + half * 16 + 8);
    u16x16 pU;
#pragma unroll
    for (int i = 0; i < 8; i++) { pU[i] = p0[i]; pU[8 + i] = p1[i]; }
    const bf16x16 pfrag = as_bf16x16(pU);
    __syncthreads();

    // P(16x32) @ V(32x128): 8 column tiles; vT rows give contiguous B-fragments.
#pragma unroll
    for (int ct = 0; ct < 8; ++ct) {
      u16x16 vb = *(const u16x16*)(vT + (size_t)(ct * 16 + lm) * NSP + kvbase + half * 16);
      acc[ct] = __builtin_amdgcn_wmma_f32_16x16x32_bf16(false, pfrag, false, as_bf16x16(vb),
                                                        (short)0, acc[ct], false, false);
    }
  }

  // Row-sum reduction across the 16 lanes of each half (matches C/D half layout).
#pragma unroll
  for (int r = 0; r < 8; r++) {
    float zz = z[r];
    zz += __shfl_xor(zz, 1, 32);  zz += __shfl_xor(zz, 2, 32);
    zz += __shfl_xor(zz, 4, 32);  zz += __shfl_xor(zz, 8, 32);
    z[r] = 1.0f / zz;
  }
#pragma unroll
  for (int ct = 0; ct < 8; ++ct)
#pragma unroll
    for (int r = 0; r < 8; r++)
      attn_out[(size_t)(qbase + half * 8 + r) * DIM + ct * 16 + lm] = acc[ct][r] * z[r];
}

// ---------------------------------------------------------------- output projection
__global__ __launch_bounds__(256) void out_proj(const float* __restrict__ attn_out,
                                                const float* __restrict__ w_out,
                                                const float* __restrict__ b_out,
                                                float* __restrict__ outf) {
  int t = blockIdx.x * 256 + threadIdx.x;
  int jcol = t & 127, n = t >> 7;
  float acc = b_out[jcol];
#pragma unroll 4
  for (int c = 0; c < DIM; ++c)
    acc += attn_out[(size_t)n * DIM + c] * w_out[c * DIM + jcol];
  outf[(size_t)n * DIM + jcol] = acc;
}

// ---------------------------------------------------------------- pixel gather (BW-bound)
__global__ __launch_bounds__(256) void gather_kernel(const int* __restrict__ seg,
                                                     const float* __restrict__ outf,
                                                     float* __restrict__ out) {
  int p = blockIdx.x * 256 + threadIdx.x;   // pixel index; coalesced stores per channel
  int s = seg[p];
  const float4* row = (const float4*)(outf + (size_t)s * DIM);  // 512B row, L2-resident
#pragma unroll
  for (int c = 0; c < 32; ++c) {
    float4 v = row[c];
    out[(size_t)(4 * c + 0) * NPIX + p] = v.x;
    out[(size_t)(4 * c + 1) * NPIX + p] = v.y;
    out[(size_t)(4 * c + 2) * NPIX + p] = v.z;
    out[(size_t)(4 * c + 3) * NPIX + p] = v.w;
  }
}

// ---------------------------------------------------------------- launch
extern "C" void kernel_launch(void* const* d_in, const int* in_sizes, int n_in,
                              void* d_out, int out_size, void* d_ws, size_t ws_size,
                              hipStream_t stream) {
  const float* spf   = (const float*)d_in[0];   // (128, 4096)
  const int*   seg   = (const int*)d_in[1];     // (512, 512)
  const float* w_qkv = (const float*)d_in[2];   // (128, 384)
  const float* w_out = (const float*)d_in[3];   // (128, 128)
  const float* b_out = (const float*)d_in[4];   // (128,)
  float* out = (float*)d_out;                   // (1,128,512,512)
  (void)in_sizes; (void)n_in; (void)out_size; (void)ws_size;

  char* ws = (char*)d_ws;
  size_t off = 0;
  auto walloc = [&](size_t bytes) -> char* {
    char* p = ws + off;
    off += (bytes + 255) & ~(size_t)255;
    return p;
  };
  unsigned*       adj      = (unsigned*)      walloc((size_t)NSP * 128 * 4); // 2 MB bitmask
  float*          q0raw    = (float*)         walloc((size_t)NSP * 32 * 4);
  float*          k0raw    = (float*)         walloc((size_t)NSP * 32 * 4);
  unsigned short* q0n      = (unsigned short*)walloc((size_t)NSP * 32 * 2);
  unsigned short* k0n      = (unsigned short*)walloc((size_t)NSP * 32 * 2);
  unsigned short* vT       = (unsigned short*)walloc((size_t)DIM * NSP * 2); // 1 MB
  float*          attn_buf = (float*)         walloc((size_t)NSP * DIM * 4);
  float*          outf     = (float*)         walloc((size_t)NSP * DIM * 4);

  zero_u32   <<<(NSP * 128 + 255) / 256, 256, 0, stream>>>(adj, NSP * 128);
  build_adj  <<<(510 * 510 + 255) / 256, 256, 0, stream>>>(seg, adj);
  qkv_kernel <<<(NSP * 192) / 256,       256, 0, stream>>>(spf, w_qkv, q0raw, k0raw, vT);
  norm_kernel<<<(2 * NSP * 32) / 256,    256, 0, stream>>>(q0raw, k0raw, q0n, k0n);
  attn_kernel<<<NSP / 16,                 32, 0, stream>>>(q0n, k0n, vT, adj, attn_buf);
  out_proj   <<<(NSP * DIM) / 256,       256, 0, stream>>>(attn_buf, w_out, b_out, outf);
  gather_kernel<<<NPIX / 256,            256, 0, stream>>>(seg, outf, out);
}
